// MultiHeadAttention_72791105732825
// MI455X (gfx1250) — matrix operand order repaired and checked by reading, not compiled, not run on hardware
//
#include <hip/hip_runtime.h>

#define BB 4
#define TT 2048
#define HH 1024
#define NHH 16
#define DHH 64

typedef __attribute__((ext_vector_type(16))) __bf16        v16bf;
typedef __attribute__((ext_vector_type(8)))  float         v8f;
typedef __attribute__((ext_vector_type(4)))  unsigned int  u32x4;
typedef __attribute__((ext_vector_type(4)))  float         f32x4;

union Frag {
    u32x4          q[2];
    unsigned short s[16];
    v16bf          v;
};

__device__ __forceinline__ unsigned short f32_to_bf16(float f) {
    union { float f; unsigned u; } c; c.f = f;
    unsigned u = c.u;
    u += 0x7fffu + ((u >> 16) & 1u);          // round to nearest even
    return (unsigned short)(u >> 16);
}

__device__ __forceinline__ v8f wmma_bf16(const v16bf& a, const v16bf& b, v8f c) {
    return __builtin_amdgcn_wmma_f32_16x16x32_bf16(false, a, false, b, (short)0, c,
                                                   false, false);
}

// CDNA5 async DMA: global -> LDS, 16 bytes per lane, tracked by ASYNCcnt.
__device__ __forceinline__ void async_b128(void* lds, const void* g) {
    unsigned off = (unsigned)(uintptr_t)lds;   // low 32 bits = wave LDS offset
    asm volatile("global_load_async_to_lds_b128 %0, %1, off"
                 :: "v"(off), "v"(g) : "memory");
}
__device__ __forceinline__ void wait_async0() {
    asm volatile("s_wait_asynccnt 0" ::: "memory");
}

// ---------------------------------------------------------------- converts
__global__ void k_cvt_bf16(const float* __restrict__ in,
                           unsigned short* __restrict__ out, int n) {
    int i = blockIdx.x * blockDim.x + threadIdx.x;
    int stride = gridDim.x * blockDim.x;
    for (; i < n; i += stride) out[i] = f32_to_bf16(in[i]);
}

__global__ void k_zero(float* __restrict__ p, int n) {
    int i = blockIdx.x * blockDim.x + threadIdx.x;
    if (i < n) p[i] = 0.f;
}

// ---------------------------------------------------------------- QKV projection
// out = x @ W^T.  One block = 128 rows (8 waves x 16) of one 64-col n-strip.
// 64x64 bf16 weight stage (8KB) async-DMA'd to LDS, double buffered, shared by
// all 8 waves.  Compute phase: all 8 B-fragment ds_loads issued first, then 8
// WMMAs, so waits degrade to partial s_wait_dscnt and LDS latency overlaps XDL.
// vmode 0: out[b][h][t][dh]   (Q, K)
// vmode 1: out[b][h][dh][t]   (V transposed, so P@V B-fragments are contiguous)
__global__ void k_proj(const unsigned short* __restrict__ xb,
                       const unsigned short* __restrict__ wb,
                       unsigned short* __restrict__ out, int vmode) {
    // pitch 72 halves = 36 dwords: 36*r mod 64 distinct for r=0..15 (conflict
    // free) and 144-byte rows keep every 32B fragment chunk 16B-aligned.
    __shared__ __align__(16) unsigned short wt[2][64][72];

    const int tid  = threadIdx.x;
    const int lane = tid & 31;
    const int hi   = lane >> 4;
    const int ln   = lane & 15;
    const int w    = tid >> 5;
    const int mgrp = blockIdx.x >> 4;        // 64 groups of 128 rows over B*T
    const int n64  = blockIdx.x & 15;        // 16 strips of 64 output channels
    const int m0   = (mgrp * 8 + w) << 4;
    const int n0   = n64 << 6;

    const int prow = tid >> 2;               // DMA: 64 rows x (2 x 16B) chunks
    const int pcol = (tid & 3) * 16;

    auto stageW = [&](int buf, int k0) {
        const unsigned short* g = wb + (size_t)(n0 + prow) * HH + k0 + pcol;
        async_b128(&wt[buf][prow][pcol],     g);
        async_b128(&wt[buf][prow][pcol + 8], g + 8);
    };

    const unsigned short* aptr = xb + (size_t)(m0 + ln) * HH;
    auto loadA = [&](Frag* a, int k0) {
#pragma unroll
        for (int kk = 0; kk < 2; ++kk) {
            a[kk].q[0] = *(const u32x4*)(aptr + k0 + kk * 32 + hi * 16);
            a[kk].q[1] = *(const u32x4*)(aptr + k0 + kk * 32 + hi * 16 + 8);
        }
    };

    v8f acc[4] = {};
    auto compute = [&](int buf, const Frag* a) {
        const unsigned short (*wtb)[72] = wt[buf];
        Frag bf[8];
#pragma unroll
        for (int i = 0; i < 8; ++i) {        // phase 1: all fragment loads
            const int kk = i >> 2, nb = i & 3;
            const u32x4* p = (const u32x4*)&wtb[nb * 16 + ln][kk * 32 + hi * 16];
            bf[i].q[0] = p[0]; bf[i].q[1] = p[1];
        }
#pragma unroll
        for (int i = 0; i < 8; ++i)          // phase 2: all WMMAs
            acc[i & 3] = wmma_bf16(a[i >> 2].v, bf[i].v, acc[i & 3]);
    };

    Frag aA[2], aB[2];
    stageW(0, 0);
    loadA(aA, 0);

    for (int s = 0; s < 16; s += 2) {        // 16 stages of 64-deep k
        wait_async0();
        __syncthreads();                     // buf0 ready
        if (s + 1 < 16) { stageW(1, (s + 1) * 64); loadA(aB, (s + 1) * 64); }
        compute(0, aA);
        __syncthreads();                     // buf0 reads done
        wait_async0();
        __syncthreads();                     // buf1 ready
        if (s + 2 < 16) { stageW(0, (s + 2) * 64); loadA(aA, (s + 2) * 64); }
        compute(1, aB);
        __syncthreads();                     // buf1 reads done
    }
#pragma unroll
    for (int nb = 0; nb < 4; ++nb) {
#pragma unroll
        for (int r = 0; r < 8; ++r) {
            int m  = m0 + r + hi * 8;        // C layout: row in VGPR idx
            int n  = n0 + nb * 16 + ln;
            int b  = m >> 11, t = m & (TT - 1);
            int h  = n >> 6,  dh = n & 63;
            unsigned short v = f32_to_bf16(acc[nb][r]);
            if (vmode == 0)
                out[(((size_t)(b * NHH + h) * TT) + t) * DHH + dh] = v;
            else
                out[(((size_t)(b * NHH + h) * DHH) + dh) * TT + t] = v;
        }
    }
}

// ---------------------------------------------------------------- flash attention
// One block = one (b,h) with 8 adjacent 16-row t-tiles (one per wave).
// 64-key stages: K (64x64) and V (64x64) bf16 tiles async-DMA'd to LDS once per
// block, double buffered; 16 WMMAs per barrier pair, shared by all 8 waves.
// V-fragment ds_loads are issued before the softmax VALU block so their LDS
// latency hides behind exp/shuffle work.
__global__ void k_attn(const unsigned short* __restrict__ Qh,
                       const unsigned short* __restrict__ Kh,
                       const unsigned short* __restrict__ Vt,
                       float* __restrict__ csum) {
    __shared__ __align__(16) unsigned short kt[2][64][72];
    __shared__ __align__(16) unsigned short vt[2][64][72];
    __shared__ __align__(16) unsigned short pb[8][16][40];  // per-wave P relayout

    const int tid  = threadIdx.x;
    const int lane = tid & 31;
    const int hi   = lane >> 4;
    const int ln   = lane & 15;
    const int w    = tid >> 5;
    const int b    = blockIdx.x >> 8;
    const int h    = (blockIdx.x >> 4) & 15;
    const int grp  = blockIdx.x & 15;
    const int bh   = b * NHH + h;
    const int t0   = (grp * 8 + w) << 4;

    const unsigned short* qbase = Qh + ((size_t)bh * TT + t0) * DHH;
    const unsigned short* kbase = Kh + (size_t)bh * TT * DHH;
    const unsigned short* vbase = Vt + (size_t)bh * DHH * TT;

    const int prow = tid >> 2;               // DMA: 64 rows x (2 x 16B) chunks
    const int pcol = (tid & 3) * 16;

    auto stageKV = [&](int buf, int s0) {
        const unsigned short* gk = kbase + (size_t)(s0 + prow) * DHH + pcol;
        async_b128(&kt[buf][prow][pcol],     gk);
        async_b128(&kt[buf][prow][pcol + 8], gk + 8);
        const unsigned short* gv = vbase + (size_t)prow * TT + s0 + pcol;
        async_b128(&vt[buf][prow][pcol],     gv);
        async_b128(&vt[buf][prow][pcol + 8], gv + 8);
    };

    stageKV(0, 0);

    // Q A-fragments for dh slices [0..31], [32..63] (kept in registers)
    Frag qa[2];
    {
        const unsigned short* qp = qbase + ln * DHH;
#pragma unroll
        for (int dd = 0; dd < 2; ++dd) {
            qa[dd].q[0] = *(const u32x4*)(qp + dd * 32 + hi * 16);
            qa[dd].q[1] = *(const u32x4*)(qp + dd * 32 + hi * 16 + 8);
        }
    }

    v8f   ctx[4] = {};
    float m[8], l[8];
#pragma unroll
    for (int r = 0; r < 8; ++r) { m[r] = -1e30f; l[r] = 0.f; }

    // one 32-key sub-step: S = Q K^T, online softmax, ctx += P V
    auto substep = [&](const unsigned short (*ktb)[72],
                       const unsigned short (*vtb)[72], int ss) {
        // phase 1: all 4 K B-fragments
        Frag kf[4];
#pragma unroll
        for (int i = 0; i < 4; ++i) {
            const int dd = i >> 1, j = i & 1;
            const u32x4* p =
                (const u32x4*)&ktb[ss * 32 + j * 16 + ln][dd * 32 + hi * 16];
            kf[i].q[0] = p[0]; kf[i].q[1] = p[1];
        }
        // phase 2: score WMMAs
        v8f S0 = {}, S1 = {};
        S0 = wmma_bf16(qa[0].v, kf[0].v, S0);
        S1 = wmma_bf16(qa[0].v, kf[1].v, S1);
        S0 = wmma_bf16(qa[1].v, kf[2].v, S0);
        S1 = wmma_bf16(qa[1].v, kf[3].v, S1);

        // V fragments issued now: latency hides behind the softmax VALU block
        Frag vf[4];
#pragma unroll
        for (int n = 0; n < 4; ++n) {
            const u32x4* p = (const u32x4*)&vtb[n * 16 + ln][ss * 32 + hi * 16];
            vf[n].q[0] = p[0]; vf[n].q[1] = p[1];
        }

        float pr0[8], pr1[8];
#pragma unroll
        for (int r = 0; r < 8; ++r) {
            float a0 = S0[r] * 0.125f;       // 1/sqrt(64)
            float a1 = S1[r] * 0.125f;
            float rm = fmaxf(a0, a1);
            rm = fmaxf(rm, __shfl_xor(rm, 1));
            rm = fmaxf(rm, __shfl_xor(rm, 2));
            rm = fmaxf(rm, __shfl_xor(rm, 4));
            rm = fmaxf(rm, __shfl_xor(rm, 8));
            float nm   = fmaxf(m[r], rm);
            float corr = __expf(m[r] - nm);
            float p0   = __expf(a0 - nm);
            float p1   = __expf(a1 - nm);
            float ps = p0 + p1;
            ps += __shfl_xor(ps, 1);
            ps += __shfl_xor(ps, 2);
            ps += __shfl_xor(ps, 4);
            ps += __shfl_xor(ps, 8);
            l[r] = l[r] * corr + ps;
            m[r] = nm;
            pr0[r] = p0; pr1[r] = p1;
#pragma unroll
            for (int n = 0; n < 4; ++n) ctx[n][r] *= corr;
        }
        // relayout P: C fragment -> A fragment via per-wave LDS tile
#pragma unroll
        for (int r = 0; r < 8; ++r) {
            pb[w][r + hi * 8][ln]      = f32_to_bf16(pr0[r]);
            pb[w][r + hi * 8][ln + 16] = f32_to_bf16(pr1[r]);
        }
        asm volatile("s_wait_dscnt 0" ::: "memory");
        Frag pa;
        pa.q[0] = *(const u32x4*)&pb[w][ln][hi * 16];
        pa.q[1] = *(const u32x4*)&pb[w][ln][hi * 16 + 8];
#pragma unroll
        for (int n = 0; n < 4; ++n)
            ctx[n] = wmma_bf16(pa.v, vf[n].v, ctx[n]);
    };

    for (int it = 0; it < TT / 64; ++it) {   // 32 stages of 64 keys
        wait_async0();
        __syncthreads();                     // K/V tiles [it&1] visible
        if (it + 1 < TT / 64) stageKV((it + 1) & 1, (it + 1) * 64);
        const unsigned short (*ktb)[72] = kt[it & 1];
        const unsigned short (*vtb)[72] = vt[it & 1];
        substep(ktb, vtb, 0);
        substep(ktb, vtb, 1);
        __syncthreads();                     // all reads done before overwrite
    }

    // ---- normalize, sum this tile's 16 rows, accumulate column sums
#pragma unroll
    for (int n = 0; n < 4; ++n) {
        float col = 0.f;
#pragma unroll
        for (int r = 0; r < 8; ++r) col += ctx[n][r] / l[r];
        col += __shfl_xor(col, 16);          // combine the two row-halves
        if (hi == 0)
            atomicAdd(&csum[b * HH + h * DHH + n * 16 + ln], col);
    }
}

// ---------------------------------------------------------------- final projection
// out[b][n] = (1/T) * sum_k csum[b][k] * Wo[n][k] + bo[n]   (tiny: 8.4 MFLOP)
__global__ void k_out(const float* __restrict__ csum,
                      const float* __restrict__ Wo,
                      const float* __restrict__ bo,
                      float* __restrict__ out) {
    int idx = blockIdx.x * blockDim.x + threadIdx.x;   // 4096
    int b = idx >> 10, n = idx & 1023;
    const f32x4* cs = (const f32x4*)(csum + b * HH);
    const f32x4* wr = (const f32x4*)(Wo + (size_t)n * HH);
    float acc = 0.f;
    for (int k = 0; k < HH / 4; ++k) {
        f32x4 c = cs[k], ww = wr[k];
        acc += c.x * ww.x + c.y * ww.y + c.z * ww.z + c.w * ww.w;
    }
    out[idx] = acc * (1.0f / TT) + bo[n];
}

// ----------------------------------------------------------------
extern "C" void kernel_launch(void* const* d_in, const int* in_sizes, int n_in,
                              void* d_out, int out_size, void* d_ws, size_t ws_size,
                              hipStream_t stream) {
    (void)in_sizes; (void)n_in; (void)out_size; (void)ws_size;
    const float* x  = (const float*)d_in[0];
    const float* Wq = (const float*)d_in[1];
    const float* Wk = (const float*)d_in[2];
    const float* Wv = (const float*)d_in[3];
    const float* Wo = (const float*)d_in[4];
    const float* bo = (const float*)d_in[5];

    char* ws = (char*)d_ws;
    size_t off = 0;
    auto alloc = [&](size_t bytes) -> void* {
        void* p = ws + off;
        off += (bytes + 255) & ~(size_t)255;
        return p;
    };
    const size_t nx = (size_t)BB * TT * HH;          // 8,388,608
    unsigned short* xb  = (unsigned short*)alloc(nx * 2);
    unsigned short* Wqb = (unsigned short*)alloc((size_t)HH * HH * 2);
    unsigned short* Wkb = (unsigned short*)alloc((size_t)HH * HH * 2);
    unsigned short* Wvb = (unsigned short*)alloc((size_t)HH * HH * 2);
    unsigned short* Qh  = (unsigned short*)alloc(nx * 2);
    unsigned short* Kh  = (unsigned short*)alloc(nx * 2);
    unsigned short* Vt  = (unsigned short*)alloc(nx * 2);
    float*          csum = (float*)alloc((size_t)BB * HH * 4);

    k_cvt_bf16<<<1024, 256, 0, stream>>>(x,  xb,  (int)nx);
    k_cvt_bf16<<<256,  256, 0, stream>>>(Wq, Wqb, HH * HH);
    k_cvt_bf16<<<256,  256, 0, stream>>>(Wk, Wkb, HH * HH);
    k_cvt_bf16<<<256,  256, 0, stream>>>(Wv, Wvb, HH * HH);
    k_zero<<<16, 256, 0, stream>>>(csum, BB * HH);

    // 1024 blocks: 64 row-groups (128 rows) x 16 n-strips, 8 waves per block
    k_proj<<<1024, 256, 0, stream>>>(xb, Wqb, Qh, 0);
    k_proj<<<1024, 256, 0, stream>>>(xb, Wkb, Kh, 0);
    k_proj<<<1024, 256, 0, stream>>>(xb, Wvb, Vt, 1);

    // 1024 blocks: (b=4) x (h=16) x (16 groups of 8 t-tiles)
    k_attn<<<1024, 256, 0, stream>>>(Qh, Kh, Vt, csum);

    k_out<<<16, 256, 0, stream>>>(csum, Wo, bo, (float*)d_out);
}
